// MaskedMultiheadSelfAttention_78709570666608
// MI455X (gfx1250) — compile-verified, matrix operand order checked by reading, not run on hardware
//
#include <hip/hip_runtime.h>

// ---------------------------------------------------------------------------
// Masked multi-head self-attention, MI455X (gfx1250, wave32, WMMA bf16)
//   B=4, T=2048, C=1024, NH=16, HD=64
// Pipeline: qkv_gemm (bf16 WMMA, V stored transposed)
//           -> flash_attn (double-buffered async LDS staging, online softmax)
//           -> out_gemm (bf16 WMMA, f32 out)
// GEMMs: block tile 128x64x32, 8 waves, 32x32 output per wave (4 WMMAs/step).
// Fragments are dword-assembled (ds_load_b128).
// ---------------------------------------------------------------------------

typedef __bf16 bf16_t;
typedef __attribute__((ext_vector_type(16))) __bf16    v16bf;
typedef __attribute__((ext_vector_type(8)))  float     v8f;
typedef __attribute__((ext_vector_type(8)))  unsigned  v8u;
typedef int v4i __attribute__((vector_size(16)));   // matches builtin's V4i

#define BATCH 4
#define SEQ   2048
#define CDIM  1024
#define NHEAD 16
#define HDIM  64
#define BT    (BATCH * SEQ)     // 8192
#define N3C   (3 * CDIM)        // 3072

#define AS1 __attribute__((address_space(1)))
#define AS3 __attribute__((address_space(3)))

#if __has_builtin(__builtin_amdgcn_global_load_async_to_lds_b128)
#define HAVE_ASYNC 1
#else
#define HAVE_ASYNC 0
#endif

// ---- helpers ---------------------------------------------------------------

static __device__ __forceinline__ unsigned short f2bf(float f) {
    unsigned u = __builtin_bit_cast(unsigned, f);
    unsigned r = u + 0x7FFFu + ((u >> 16) & 1u);   // round-to-nearest-even
    return (unsigned short)(r >> 16);
}
// pack two f32 -> dword of two bf16 (lo at low 16 bits = lower address)
static __device__ __forceinline__ unsigned pack_bf16(float lo, float hi) {
#if __has_builtin(__builtin_amdgcn_cvt_pk_bf16_f32)
    auto p = __builtin_amdgcn_cvt_pk_bf16_f32(lo, hi);
    return __builtin_bit_cast(unsigned, p);
#else
    return (unsigned)f2bf(lo) | ((unsigned)f2bf(hi) << 16);
#endif
}
static __device__ __forceinline__ v8f vzero() {
    v8f z = {0.f, 0.f, 0.f, 0.f, 0.f, 0.f, 0.f, 0.f};
    return z;
}
static __device__ __forceinline__ v8f wmma_bf16(v16bf a, v16bf b, v8f c) {
    return __builtin_amdgcn_wmma_f32_16x16x32_bf16(
        false, a, false, b, (short)0, c, false, false);
}
// 16-bit A/B fragment (16x32): VGPR i holds k-pair; dword index within a
// k-contiguous row of bf16 (ISA 7.12.2): dw = (i<4 ? i : i+4) + 4*lane_hi
static __device__ __forceinline__ v16bf load_frag_row(const unsigned* rowp,
                                                      int lane_hi) {
    v8u u;
#pragma unroll
    for (int i = 0; i < 8; ++i) {
        const int dw = ((i < 4) ? i : (i + 4)) + 4 * lane_hi;
        u[i] = rowp[dw];
    }
    return __builtin_bit_cast(v16bf, u);
}
// 16B global->LDS copy: async path on gfx1250, uint4 fallback
static __device__ __forceinline__ void cp16(void* dst_lds, const void* src_glob) {
#if HAVE_ASYNC
    __builtin_amdgcn_global_load_async_to_lds_b128(
        (AS1 v4i*)src_glob, (AS3 v4i*)dst_lds, 0, 0);
#else
    *(uint4*)dst_lds = *(const uint4*)src_glob;
#endif
}
static __device__ __forceinline__ void async_wait() {
#if HAVE_ASYNC
    asm volatile("s_wait_asynccnt 0x0" ::: "memory");
#endif
}

// ---------------------------------------------------------------------------
// Kernel 1: QKV projection. X[8192,1024] f32 x W[1024,3072] f32 + b ->
// Q,K bf16 [B,NH,T,HD]; V bf16 TRANSPOSED [B,NH,HD,T].
// Block 256 thr (8 waves), tile 128(M)x64(N)x32(K); waves 4 row x 2 col,
// each wave a 32x32 output tile (2 A-frags x 2 B-frags, 4 WMMAs per step).
// LDS: A [m][k-pair] 8KB, B transposed [n][k-pair] 4KB, packed-bf16 dwords.
// ---------------------------------------------------------------------------
__global__ void qkv_gemm(const float* __restrict__ X,
                         const float* __restrict__ W,
                         const float* __restrict__ bias,
                         unsigned short* __restrict__ Qb,
                         unsigned short* __restrict__ Kb,
                         unsigned short* __restrict__ Vtb) {
    __shared__ __align__(16) unsigned Alds[128][16];  // [m][k-pair]
    __shared__ __align__(16) unsigned Btld[64][16];   // [n][k-pair] transposed

    const int tid = threadIdx.x;
    const int lane = tid & 31, w = tid >> 5;
    const int lane_lo = lane & 15, lane_hi = (lane >> 4) & 1;
    const int wr = w >> 1, wc = w & 1;
    const int bm = blockIdx.y * 128;
    const int bn = blockIdx.x * 64;

    const int arow = tid >> 1, acol2 = (tid & 1) * 8;  // A: 16 f32 -> 8 dwords
    const int bcol = tid & 63, bk0 = (tid >> 6) * 8;   // B: 8 f32 -> 4 dwords

    v8f acc[2][2];
#pragma unroll
    for (int mt = 0; mt < 2; ++mt)
#pragma unroll
        for (int sub = 0; sub < 2; ++sub) acc[mt][sub] = vzero();

    for (int kk = 0; kk < CDIM; kk += 32) {
        const float* ap = X + (size_t)(bm + arow) * CDIM + kk + acol2 * 2;
        if (kk + 32 < CDIM)
            __builtin_prefetch(ap + 32, 0, 0);         // global_prefetch_b8
#pragma unroll
        for (int jj = 0; jj < 8; ++jj)
            Alds[arow][acol2 + jj] = pack_bf16(ap[2 * jj], ap[2 * jj + 1]);
        const float* bp = W + (size_t)(kk + bk0) * N3C + bn + bcol;
#pragma unroll
        for (int jj = 0; jj < 4; ++jj)                 // coalesced row reads
            Btld[bcol][bk0 / 2 + jj] =
                pack_bf16(bp[(size_t)(2 * jj) * N3C], bp[(size_t)(2 * jj + 1) * N3C]);
        __syncthreads();

        v16bf af[2], bf[2];
#pragma unroll
        for (int mt = 0; mt < 2; ++mt)
            af[mt] = load_frag_row(&Alds[wr * 32 + mt * 16 + lane_lo][0], lane_hi);
#pragma unroll
        for (int sub = 0; sub < 2; ++sub)
            bf[sub] = load_frag_row(&Btld[wc * 32 + sub * 16 + lane_lo][0], lane_hi);
#pragma unroll
        for (int mt = 0; mt < 2; ++mt)
#pragma unroll
            for (int sub = 0; sub < 2; ++sub)
                acc[mt][sub] = wmma_bf16(af[mt], bf[sub], acc[mt][sub]);
        __syncthreads();
    }

    // epilogue: Q,K -> [B,NH,T,HD]; V -> [B,NH,HD,T]  (which is block-uniform)
    const int which = bn >> 10;                        // 0=Q 1=K 2=V
    float bv[2];
#pragma unroll
    for (int sub = 0; sub < 2; ++sub)
        bv[sub] = bias[bn + wc * 32 + sub * 16 + lane_lo];

    if (which == 2) {                                  // V, transposed store
#pragma unroll
        for (int mt = 0; mt < 2; ++mt)
#pragma unroll
            for (int sub = 0; sub < 2; ++sub)
#pragma unroll
                for (int r = 0; r < 8; ++r) {
                    const int i = bm + wr * 32 + mt * 16 + r + 8 * lane_hi;
                    const int j = bn + wc * 32 + sub * 16 + lane_lo;
                    const int cc = j & (CDIM - 1);
                    const int h = cc >> 6, d = cc & (HDIM - 1);
                    const int bb = i >> 11, t = i & (SEQ - 1);
                    Vtb[((size_t)(bb * NHEAD + h) * HDIM + d) * SEQ + t] =
                        f2bf(acc[mt][sub][r] + bv[sub]);
                }
    } else {                                           // Q or K, row-major
        unsigned short* base = (which == 0) ? Qb : Kb;
#pragma unroll
        for (int mt = 0; mt < 2; ++mt)
#pragma unroll
            for (int sub = 0; sub < 2; ++sub)
#pragma unroll
                for (int r = 0; r < 8; ++r) {
                    const int i = bm + wr * 32 + mt * 16 + r + 8 * lane_hi;
                    const int j = bn + wc * 32 + sub * 16 + lane_lo;
                    const int cc = j & (CDIM - 1);
                    const int h = cc >> 6, d = cc & (HDIM - 1);
                    const int bb = i >> 11, t = i & (SEQ - 1);
                    base[((size_t)(bb * NHEAD + h) * SEQ + t) * HDIM + d] =
                        f2bf(acc[mt][sub][r] + bv[sub]);
                }
    }
}

// ---------------------------------------------------------------------------
// Kernel 2: flash attention (causal + additive pad mask), bf16 WMMA.
// Block 128 thr (4 waves) owns 64 query rows of one (b,h); each wave owns 16.
// Double-buffered async staging: while computing kv step s from LDS buffer
// (s&1), step s+1's K tile [32kv][64d] + Vt tile [64d][32kv] stream into the
// other buffer via global_load_async_to_lds_b128; s_wait_asynccnt 0 + barrier
// after compute.
// ---------------------------------------------------------------------------
__global__ void flash_attn(const unsigned short* __restrict__ Qb,
                           const unsigned short* __restrict__ Kb,
                           const unsigned short* __restrict__ Vtb,
                           const float* __restrict__ amask,
                           unsigned short* __restrict__ WVb) {
    __shared__ __align__(16) unsigned short Klds[2][32][64];    // [buf][kv][d]
    __shared__ __align__(16) unsigned short Vtlds[2][64][32];   // [buf][d][kv]
    __shared__ __align__(16) unsigned short Plds[4][16][32];    // per-wave P

    const int tid = threadIdx.x;
    const int lane = tid & 31, w = tid >> 5;
    const int lane_lo = lane & 15, lane_hi = (lane >> 4) & 1;
    const int bh = blockIdx.x;
    const int b = bh >> 4;
    const int h = bh & (NHEAD - 1);
    const int q0b = blockIdx.y * 64;
    const int q0 = q0b + w * 16;

    const unsigned short* Qh  = Qb  + (size_t)bh * SEQ * HDIM;
    const unsigned short* Kh  = Kb  + (size_t)bh * SEQ * HDIM;
    const unsigned short* VTh = Vtb + (size_t)bh * HDIM * SEQ;

    auto stage = [&](int s, int pb) {
        const int kv0 = s << 5;
#pragma unroll
        for (int c = 0; c < 2; ++c) {
            const int idx = tid + c * 128;             // 0..255
            const int kr = idx >> 3, kc = (idx & 7) * 8;
            cp16(&Klds[pb][kr][kc], Kh + (size_t)(kv0 + kr) * HDIM + kc);
            const int vr = idx >> 2, vc = (idx & 3) * 8;
            cp16(&Vtlds[pb][vr][vc], VTh + (size_t)vr * SEQ + kv0 + vc);
        }
    };

    v16bf qf[2];
    {
        const unsigned* qrow =
            (const unsigned*)(Qh + (size_t)(q0 + lane_lo) * HDIM);
        qf[0] = load_frag_row(qrow, lane_hi);
        qf[1] = load_frag_row(qrow + 16, lane_hi);
    }

    v8f o[4];
#pragma unroll
    for (int nt = 0; nt < 4; ++nt) o[nt] = vzero();
    float mst[8], lst[8];
#pragma unroll
    for (int r = 0; r < 8; ++r) { mst[r] = -3.0e38f; lst[r] = 0.f; }

    const float scale = 0.125f;                    // 1/sqrt(64)
    const int nsteps = (q0b + 64 + 31) >> 5;       // block-level causal bound

    stage(0, 0);
    async_wait();
    __syncthreads();

    for (int s = 0; s < nsteps; ++s) {
        const int pb = s & 1;
        const int kv0 = s << 5;

        if (s + 1 < nsteps) stage(s + 1, pb ^ 1);  // overlap with compute

        if (kv0 <= q0 + 15) {                      // this wave's causal limit
            // ---- S = Q * K^T (two 16x16 score tiles) ----
            v8f sc[2];
            float pad[2];
#pragma unroll
            for (int nt = 0; nt < 2; ++nt) {
                const int n16 = nt * 16 + lane_lo;
                const unsigned* krow = (const unsigned*)&Klds[pb][n16][0];
                const v16bf kb0 = load_frag_row(krow, lane_hi);
                const v16bf kb1 = load_frag_row(krow + 16, lane_hi);
                v8f t = vzero();
                t = wmma_bf16(qf[0], kb0, t);
                t = wmma_bf16(qf[1], kb1, t);
                sc[nt] = t;
                pad[nt] = (1.0f - amask[b * SEQ + kv0 + n16]) * (-3.0e38f);
            }

            // ---- causal + pad mask, scale ----
#pragma unroll
            for (int nt = 0; nt < 2; ++nt) {
                const int n = kv0 + nt * 16 + lane_lo;
#pragma unroll
                for (int r = 0; r < 8; ++r) {
                    const int m = q0 + r + 8 * lane_hi;
                    float v = sc[nt][r] * scale + pad[nt];
                    if (n > m) v = -3.0e38f;
                    sc[nt][r] = v;
                }
            }

            // ---- online softmax (rows live across 16-lane halves) ----
            float alpha[8];
#pragma unroll
            for (int r = 0; r < 8; ++r) {
                float mx = fmaxf(sc[0][r], sc[1][r]);
                mx = fmaxf(mx, __shfl_xor(mx, 1, 32));
                mx = fmaxf(mx, __shfl_xor(mx, 2, 32));
                mx = fmaxf(mx, __shfl_xor(mx, 4, 32));
                mx = fmaxf(mx, __shfl_xor(mx, 8, 32));
                const float mnew = fmaxf(mst[r], mx);
                const float a = __expf(mst[r] - mnew);
                const float e0 = __expf(sc[0][r] - mnew);
                const float e1 = __expf(sc[1][r] - mnew);
                float rs = e0 + e1;
                rs += __shfl_xor(rs, 1, 32);
                rs += __shfl_xor(rs, 2, 32);
                rs += __shfl_xor(rs, 4, 32);
                rs += __shfl_xor(rs, 8, 32);
                lst[r] = lst[r] * a + rs;
                mst[r] = mnew;
                alpha[r] = a;
                const int m = r + 8 * lane_hi;
                Plds[w][m][lane_lo]      = f2bf(e0);
                Plds[w][m][16 + lane_lo] = f2bf(e1);
            }
#pragma unroll
            for (int nt = 0; nt < 4; ++nt)
#pragma unroll
                for (int r = 0; r < 8; ++r) o[nt][r] *= alpha[r];

            asm volatile("s_wait_dscnt 0" ::: "memory");  // wave-local P bounce

            const v16bf pf =
                load_frag_row((const unsigned*)&Plds[w][lane_lo][0], lane_hi);

            // ---- O += P * V ----
#pragma unroll
            for (int nt = 0; nt < 4; ++nt) {
                const v16bf vf = load_frag_row(
                    (const unsigned*)&Vtlds[pb][nt * 16 + lane_lo][0], lane_hi);
                o[nt] = wmma_bf16(pf, vf, o[nt]);
            }
        }

        async_wait();      // next tile landed in pb^1
        __syncthreads();   // all waves done reading pb / next tile visible
    }

    // ---- epilogue: O /= l, write bf16 WV in [B*T, C] ----
#pragma unroll
    for (int nt = 0; nt < 4; ++nt) {
#pragma unroll
        for (int r = 0; r < 8; ++r) {
            const int m = q0 + r + 8 * lane_hi;
            const int col = h * HDIM + nt * 16 + lane_lo;
            WVb[(size_t)(b * SEQ + m) * CDIM + col] = f2bf(o[nt][r] / lst[r]);
        }
    }
}

// ---------------------------------------------------------------------------
// Kernel 3: output projection. WV[8192,1024] bf16 x W_out[1024,1024] f32 +
// b_out -> f32 out. Same 128x64x32 tiling; A tile async-copied to LDS.
// ---------------------------------------------------------------------------
__global__ void out_gemm(const unsigned short* __restrict__ A,
                         const float* __restrict__ W,
                         const float* __restrict__ bias,
                         float* __restrict__ Out) {
    __shared__ __align__(16) unsigned Alds[128][16];  // [m][k-pair]
    __shared__ __align__(16) unsigned Btld[64][16];   // [n][k-pair] transposed

    const int tid = threadIdx.x;
    const int lane = tid & 31, w = tid >> 5;
    const int lane_lo = lane & 15, lane_hi = (lane >> 4) & 1;
    const int wr = w >> 1, wc = w & 1;
    const int bm = blockIdx.y * 128;
    const int bn = blockIdx.x * 64;

    const int bcol = tid & 63, bk0 = (tid >> 6) * 8;  // B: 8 f32 -> 4 dwords

    v8f acc[2][2];
#pragma unroll
    for (int mt = 0; mt < 2; ++mt)
#pragma unroll
        for (int sub = 0; sub < 2; ++sub) acc[mt][sub] = vzero();

    for (int kk = 0; kk < CDIM; kk += 32) {
        // A tile 128x32 bf16 = 8KB: 2 async 16B chunks per thread
#pragma unroll
        for (int c = 0; c < 2; ++c) {
            const int idx = tid + c * 256;             // 0..511
            const int ar = idx >> 2, ac = (idx & 3) * 8;
            cp16(&Alds[ar][ac / 2], A + (size_t)(bm + ar) * CDIM + kk + ac);
        }
        const float* bp = W + (size_t)(kk + bk0) * CDIM + bn + bcol;
#pragma unroll
        for (int jj = 0; jj < 4; ++jj)
            Btld[bcol][bk0 / 2 + jj] =
                pack_bf16(bp[(size_t)(2 * jj) * CDIM], bp[(size_t)(2 * jj + 1) * CDIM]);
        async_wait();
        __syncthreads();

        v16bf af[2], bf[2];
#pragma unroll
        for (int mt = 0; mt < 2; ++mt)
            af[mt] = load_frag_row(&Alds[wr * 32 + mt * 16 + lane_lo][0], lane_hi);
#pragma unroll
        for (int sub = 0; sub < 2; ++sub)
            bf[sub] = load_frag_row(&Btld[wc * 32 + sub * 16 + lane_lo][0], lane_hi);
#pragma unroll
        for (int mt = 0; mt < 2; ++mt)
#pragma unroll
            for (int sub = 0; sub < 2; ++sub)
                acc[mt][sub] = wmma_bf16(af[mt], bf[sub], acc[mt][sub]);
        __syncthreads();
    }

    float bv[2];
#pragma unroll
    for (int sub = 0; sub < 2; ++sub)
        bv[sub] = bias[bn + wc * 32 + sub * 16 + lane_lo];
#pragma unroll
    for (int mt = 0; mt < 2; ++mt)
#pragma unroll
        for (int sub = 0; sub < 2; ++sub)
#pragma unroll
            for (int r = 0; r < 8; ++r) {
                const int i = bm + wr * 32 + mt * 16 + r + 8 * lane_hi;
                const int j = bn + wc * 32 + sub * 16 + lane_lo;
                Out[(size_t)i * CDIM + j] = acc[mt][sub][r] + bv[sub];
            }
}

// ---------------------------------------------------------------------------
// Launch
// ---------------------------------------------------------------------------
extern "C" void kernel_launch(void* const* d_in, const int* in_sizes, int n_in,
                              void* d_out, int out_size, void* d_ws, size_t ws_size,
                              hipStream_t stream) {
    const float* x     = (const float*)d_in[0];   // (B,T,C)
    const float* amask = (const float*)d_in[1];   // (B,T)
    const float* Wqkv  = (const float*)d_in[2];   // (C,3C)
    const float* bqkv  = (const float*)d_in[3];   // (3C,)
    const float* Wout  = (const float*)d_in[4];   // (C,C)
    const float* bout  = (const float*)d_in[5];   // (C,)
    float* out = (float*)d_out;                   // (B,T,C) f32

    const size_t tensorElems = (size_t)BATCH * NHEAD * SEQ * HDIM;  // 8M
    unsigned short* Qb  = (unsigned short*)d_ws;
    unsigned short* Kb  = Qb + tensorElems;
    unsigned short* Vtb = Kb + tensorElems;       // V transposed [B,NH,HD,T]
    unsigned short* WVb = Vtb + tensorElems;      // [B*T, C] bf16

    dim3 g1(N3C / 64, BT / 128);                  // (48, 64)
    qkv_gemm<<<g1, 256, 0, stream>>>(x, Wqkv, bqkv, Qb, Kb, Vtb);

    dim3 g2(BATCH * NHEAD, SEQ / 64);             // (64, 32)
    flash_attn<<<g2, 128, 0, stream>>>(Qb, Kb, Vtb, amask, WVb);

    dim3 g3(CDIM / 64, BT / 128);                 // (16, 64)
    out_gemm<<<g3, 256, 0, stream>>>(WVb, Wout, bout, out);
}
